// MoEResNetBKLayer_9002251452583
// MI455X (gfx1250) — compile-verified, matrix operand order checked by reading, not run on hardware
//
#include <hip/hip_runtime.h>
#include <hip/hip_bf16.h>

// ---------------- problem constants ----------------
constexpr int Bq = 4, Nq = 4096, Dq = 512, Eq = 4, Hq = 2048;
constexpr int T  = Bq * Nq;          // 16384 tokens
constexpr int MT = 32;               // tokens per FFN block
constexpr int HC = 256;              // H chunk per pipeline stage
constexpr float V_MAX = 3.0f, FEAT_CLAMP = 10.0f;

typedef __attribute__((ext_vector_type(16))) _Float16 v16h;
typedef __attribute__((ext_vector_type(8)))  _Float16 v8h;
typedef __attribute__((ext_vector_type(8)))  float    v8f;
typedef __attribute__((ext_vector_type(4)))  float    v4f;

// ---------------- kernel 0: zero expert counters ----------------
__global__ void zero_counts_kernel(int* counts) {
    if (threadIdx.x < Eq) counts[threadIdx.x] = 0;
}

// ---------------- kernel 1: transpose + f32->f16 convert ----------------
// dst[C][R] = (f16) src[R][C], one expert slice per blockIdx.z
__global__ __launch_bounds__(256) void transpose_cvt_kernel(
    const float* __restrict__ src, _Float16* __restrict__ dst, int R, int C) {
    __shared__ float tile[32][33];
    const size_t eoff = (size_t)blockIdx.z * R * C;
    src += eoff; dst += eoff;
    const int r0 = blockIdx.y * 32, c0 = blockIdx.x * 32;
    const int tx = threadIdx.x, ty = threadIdx.y;          // block (32,8)
#pragma unroll
    for (int i = 0; i < 32; i += 8)
        tile[ty + i][tx] = src[(size_t)(r0 + ty + i) * C + c0 + tx];
    __syncthreads();
#pragma unroll
    for (int i = 0; i < 32; i += 8)
        dst[(size_t)(c0 + ty + i) * R + r0 + tx] = (_Float16)tile[tx][ty + i];
}

// ---------------- kernel 1b: x f32 -> f16 (row-major copy) ----------------
__global__ __launch_bounds__(256) void cvt_x_kernel(
    const float* __restrict__ x, _Float16* __restrict__ xh) {
    const size_t i = ((size_t)blockIdx.x * 256 + threadIdx.x) * 8;
    const v4f a = *(const v4f*)(x + i);
    const v4f b = *(const v4f*)(x + i + 4);
    v8h o;
    o[0] = (_Float16)a.x; o[1] = (_Float16)a.y; o[2] = (_Float16)a.z; o[3] = (_Float16)a.w;
    o[4] = (_Float16)b.x; o[5] = (_Float16)b.y; o[6] = (_Float16)b.z; o[7] = (_Float16)b.w;
    *(v8h*)(xh + i) = o;
}

// ---------------- kernel 2: gate + top-1 routing ----------------
__global__ __launch_bounds__(256) void gate_kernel(
    const float* __restrict__ x, const float* __restrict__ gw,
    const float* __restrict__ gb, int* __restrict__ lists, int* __restrict__ counts) {
    const int tok  = (blockIdx.x * blockDim.x + threadIdx.x) >> 5;
    const int lane = threadIdx.x & 31;
    if (tok >= T) return;
    const float* xr = x + (size_t)tok * Dq;
    float acc0 = 0.f, acc1 = 0.f, acc2 = 0.f, acc3 = 0.f;
    for (int d = lane; d < Dq; d += 32) {
        const float xv = xr[d];
        const v4f g = *(const v4f*)(gw + d * 4);
        acc0 += xv * g.x; acc1 += xv * g.y; acc2 += xv * g.z; acc3 += xv * g.w;
    }
#pragma unroll
    for (int off = 16; off >= 1; off >>= 1) {
        acc0 += __shfl_xor(acc0, off, 32);
        acc1 += __shfl_xor(acc1, off, 32);
        acc2 += __shfl_xor(acc2, off, 32);
        acc3 += __shfl_xor(acc3, off, 32);
    }
    if (lane == 0) {
        float lg[4] = { acc0 + gb[0], acc1 + gb[1], acc2 + gb[2], acc3 + gb[3] };
        int best = 0; float bv = lg[0];
#pragma unroll
        for (int e = 1; e < Eq; ++e)          // strictly-greater => lowest index on ties (top_k)
            if (lg[e] > bv) { bv = lg[e]; best = e; }
        const int pos = atomicAdd(&counts[best], 1);
        lists[best * T + pos] = tok;
    }
}

// ---------------- kernel 3: expert FFN via WMMA ----------------
__device__ __forceinline__ float gelu_tanh(float v) {
    const float u = 0.7978845608028654f * (v + 0.044715f * v * v * v);
    return 0.5f * v * (1.0f + tanhf(u));
}

__device__ __forceinline__ v16h cat16(v8h lo, v8h hi) {
    return __builtin_shufflevector(lo, hi, 0,1,2,3,4,5,6,7,8,9,10,11,12,13,14,15);
}

__global__ __launch_bounds__(256) void ffn_kernel(
    const _Float16* __restrict__ xh,    // [T][D] f16
    const _Float16* __restrict__ w1t,   // [E][H][D] f16
    const float* __restrict__ b1,       // [E][H]
    const _Float16* __restrict__ w2t,   // [E][D][H] f16
    const float* __restrict__ b2,       // [E][D]
    const int* __restrict__ lists, const int* __restrict__ counts,
    float* __restrict__ out) {
    __shared__ _Float16 s_x[MT][Dq + 8];   // padded: conflict-free b128 reads
    __shared__ _Float16 s_h[MT][HC + 8];
    __shared__ int s_tok[MT];

    const int e    = blockIdx.y;
    const int cnt  = counts[e];
    const int base = blockIdx.x * MT;
    if (base >= cnt) return;                       // block-uniform exit (EXEC stays full)

    const int tid = threadIdx.x, lane = tid & 31, w = tid >> 5;
    const int lm = lane & 15, hi = lane >> 4;

    if (tid < MT) {
        const int idx = base + tid;
        s_tok[tid] = (idx < cnt) ? lists[e * T + idx] : -1;
    }
    __syncthreads();

    { // stage x tile (32 rows x 512 f16) into LDS via async load-to-LDS (ASYNCcnt path)
        const int row = tid >> 3;
        const int cb  = (tid & 7) * 64;            // 64 halves = 128 bytes per thread
        const int tk  = s_tok[row];
        const _Float16* src = xh + (size_t)(tk < 0 ? 0 : tk) * Dq + cb;
        unsigned lds = (unsigned)(uintptr_t)&s_x[row][cb];
        unsigned long long ga = (unsigned long long)(uintptr_t)src;
#pragma unroll
        for (int j = 0; j < 8; ++j) {              // 8 x b128 = 128B per lane
            unsigned l = lds + j * 16;
            unsigned long long g = ga + j * 16;
            asm volatile("global_load_async_to_lds_b128 %0, %1, off"
                         :: "v"(l), "v"(g) : "memory");
        }
        asm volatile("s_wait_asynccnt 0x0" ::: "memory");
    }
    __syncthreads();

    v8f Y[2][4];                                   // 32 rows x 64 cols per wave
#pragma unroll
    for (int mt = 0; mt < 2; ++mt)
#pragma unroll
        for (int nt = 0; nt < 4; ++nt) Y[mt][nt] = (v8f)0.0f;

    for (int hc = 0; hc < Hq; hc += HC) {
        // ---- phase 1: h_chunk = gelu(X @ W1[:, hc..hc+HC) + b1) ; wave owns 32 cols ----
        v8f hacc[2][2];
#pragma unroll
        for (int mt = 0; mt < 2; ++mt)
#pragma unroll
            for (int nt = 0; nt < 2; ++nt) hacc[mt][nt] = (v8f)0.0f;

        for (int k0 = 0; k0 < Dq; k0 += 32) {
            v16h a[2];
#pragma unroll
            for (int mt = 0; mt < 2; ++mt) {
                const v8h lo = *(const v8h*)&s_x[mt * 16 + lm][k0 + 8 * hi];
                const v8h hh = *(const v8h*)&s_x[mt * 16 + lm][k0 + 16 + 8 * hi];
                a[mt] = cat16(lo, hh);
            }
#pragma unroll
            for (int nt = 0; nt < 2; ++nt) {
                const int col = hc + w * 32 + nt * 16 + lm;
                const _Float16* bp = w1t + ((size_t)(e * Hq + col)) * Dq + k0 + 8 * hi;
                const v16h b = cat16(*(const v8h*)bp, *(const v8h*)(bp + 16));
#pragma unroll
                for (int mt = 0; mt < 2; ++mt)
                    hacc[mt][nt] = __builtin_amdgcn_wmma_f32_16x16x32_f16(
                        false, a[mt], false, b, (short)0, hacc[mt][nt], false, false);
            }
        }
        __syncthreads();                            // prior phase-2 readers done
#pragma unroll
        for (int nt = 0; nt < 2; ++nt) {
            const int col  = hc + w * 32 + nt * 16 + lm;
            const float bias = b1[e * Hq + col];
#pragma unroll
            for (int mt = 0; mt < 2; ++mt)
#pragma unroll
                for (int r = 0; r < 8; ++r) {
                    const float g = gelu_tanh(hacc[mt][nt][r] + bias);
                    s_h[mt * 16 + r + 8 * hi][w * 32 + nt * 16 + lm] = (_Float16)g;
                }
        }
        __syncthreads();

        // ---- phase 2: Y += h_chunk(32x256) @ W2ᵀ chunk ----
        for (int k0 = 0; k0 < HC; k0 += 32) {
            v16h a[2];
#pragma unroll
            for (int mt = 0; mt < 2; ++mt) {
                const v8h lo = *(const v8h*)&s_h[mt * 16 + lm][k0 + 8 * hi];
                const v8h hh = *(const v8h*)&s_h[mt * 16 + lm][k0 + 16 + 8 * hi];
                a[mt] = cat16(lo, hh);
            }
#pragma unroll
            for (int nt = 0; nt < 4; ++nt) {
                const int n = w * 64 + nt * 16 + lm;
                const _Float16* bp = w2t + ((size_t)(e * Dq + n)) * Hq + hc + k0 + 8 * hi;
                const v16h b = cat16(*(const v8h*)bp, *(const v8h*)(bp + 16));
#pragma unroll
                for (int mt = 0; mt < 2; ++mt)
                    Y[mt][nt] = __builtin_amdgcn_wmma_f32_16x16x32_f16(
                        false, a[mt], false, b, (short)0, Y[mt][nt], false, false);
            }
        }
    }

    // ---- epilogue: + b2, scatter rows to out ----
#pragma unroll
    for (int nt = 0; nt < 4; ++nt) {
        const int col  = w * 64 + nt * 16 + lm;
        const float bias = b2[e * Dq + col];
#pragma unroll
        for (int mt = 0; mt < 2; ++mt)
#pragma unroll
            for (int r = 0; r < 8; ++r) {
                const int tk = s_tok[mt * 16 + r + 8 * hi];
                if (tk >= 0) out[(size_t)tk * Dq + col] = Y[mt][nt][r] + bias;
            }
    }
}

// ---------------- kernel 4: v = clip(ffn . v_w + v_b) ----------------
__global__ __launch_bounds__(256) void v_kernel(
    const float* __restrict__ ffn, const float* __restrict__ v_w,
    const float* __restrict__ v_b, float* __restrict__ v) {
    const int tok  = (blockIdx.x * blockDim.x + threadIdx.x) >> 5;
    const int lane = threadIdx.x & 31;
    if (tok >= T) return;
    const float* r = ffn + (size_t)tok * Dq;
    float acc = 0.f;
    for (int d = lane; d < Dq; d += 32) acc += r[d] * v_w[d];
#pragma unroll
    for (int off = 16; off >= 1; off >>= 1) acc += __shfl_xor(acc, off, 32);
    if (lane == 0)
        v[tok] = fminf(fmaxf(acc + v_b[0], -V_MAX), V_MAX);
}

// ---------------- kernel 5: Moebius-matrix segmented scan for both continuants ----
// L_i = d_i - 1/L_{i-1} == 2x2 complex matrix chain [[d,-1],[1,0]] (scale-invariant).
// 8 chains (L/R x 4 batches) x 128 segments x 32 steps; serial depth 32+128+32.
__global__ __launch_bounds__(1024) void tridiag_scan_kernel(
    const float* __restrict__ v,
    float* __restrict__ Lr, float* __restrict__ Li,
    float* __restrict__ Rr, float* __restrict__ Ri) {
    __shared__ float sm[1024][8];                 // per-segment 2x2 complex matrix
    const int tid   = threadIdx.x;
    const int chain = tid >> 7;                   // 0..7
    const int seg   = tid & 127;                  // 0..127
    const int b     = chain >> 1;
    const int isR   = chain & 1;
    const float* vb = v + b * Nq;

    // ---- phase A: segment matrix product (normalized) ----
    float ar = 1.f, ai = 0.f, br = 0.f, bi = 0.f, cr = 0.f, ci = 0.f, dr = 1.f, di = 0.f;
    for (int j = 0; j < 32; ++j) {
        const int p   = seg * 32 + j;
        const int idx = isR ? (Nq - 1 - p) : p;
        const float er = vb[idx] - 2.0f, ei = -1.0f;      // d = (v-2) - 1j
        // P = M(e) * P with M = [[e,-1],[1,0]]
        const float nar = er * ar - ei * ai - cr;
        const float nai = er * ai + ei * ar - ci;
        const float nbr = er * br - ei * bi - dr;
        const float nbi = er * bi + ei * br - di;
        cr = ar; ci = ai; dr = br; di = bi;
        ar = nar; ai = nai; br = nbr; bi = nbi;
    }
    {
        float mx = fmaxf(fmaxf(fabsf(ar), fabsf(ai)), fmaxf(fabsf(br), fabsf(bi)));
        mx = fmaxf(mx, fmaxf(fmaxf(fabsf(cr), fabsf(ci)), fmaxf(fabsf(dr), fabsf(di))));
        const float s = __builtin_amdgcn_rcpf(fmaxf(mx, 1e-30f));
        ar *= s; ai *= s; br *= s; bi *= s; cr *= s; ci *= s; dr *= s; di *= s;
    }
    float* m = sm[chain * 128 + seg];
    m[0] = ar; m[1] = ai; m[2] = br; m[3] = bi; m[4] = cr; m[5] = ci; m[6] = dr; m[7] = di;
    __syncthreads();

    // ---- phase B: serial inclusive scan of 128 segment matrices per chain ----
    if (seg == 0) {
        float par = m[0], pai = m[1], pbr = m[2], pbi = m[3];
        float pcr = m[4], pci = m[5], pdr = m[6], pdi = m[7];
        for (int ss = 1; ss < 128; ++ss) {
            float* q = sm[chain * 128 + ss];
            const float sar = q[0], sai = q[1], sbr = q[2], sbi = q[3];
            const float scr = q[4], sci = q[5], sdr = q[6], sdi = q[7];
            // P = S * Pprev   (2x2 complex)
            const float nar = sar * par - sai * pai + sbr * pcr - sbi * pci;
            const float nai = sar * pai + sai * par + sbr * pci + sbi * pcr;
            const float nbr = sar * pbr - sai * pbi + sbr * pdr - sbi * pdi;
            const float nbi = sar * pbi + sai * pbr + sbr * pdi + sbi * pdr;
            const float ncr = scr * par - sci * pai + sdr * pcr - sdi * pci;
            const float nci = scr * pai + sci * par + sdr * pci + sdi * pcr;
            const float ndr = scr * pbr - sci * pbi + sdr * pdr - sdi * pdi;
            const float ndi = scr * pbi + sci * pbr + sdr * pdi + sdi * pdr;
            float mx = fmaxf(fmaxf(fabsf(nar), fabsf(nai)), fmaxf(fabsf(nbr), fabsf(nbi)));
            mx = fmaxf(mx, fmaxf(fmaxf(fabsf(ncr), fabsf(nci)), fmaxf(fabsf(ndr), fabsf(ndi))));
            const float s = __builtin_amdgcn_rcpf(fmaxf(mx, 1e-30f));
            par = nar * s; pai = nai * s; pbr = nbr * s; pbi = nbi * s;
            pcr = ncr * s; pci = nci * s; pdr = ndr * s; pdi = ndi * s;
            q[0] = par; q[1] = pai; q[2] = pbr; q[3] = pbi;
            q[4] = pcr; q[5] = pci; q[6] = pdr; q[7] = pdi;
        }
    }
    __syncthreads();

    // ---- phase C: regenerate 32 outputs from boundary state (homogeneous form) ----
    float pr, pi, qr, qi;                          // state (p,q), L = p/q
    if (seg == 0) { pr = 1.f; pi = 0.f; qr = 0.f; qi = 0.f; }
    else {
        const float* q = sm[chain * 128 + seg - 1];
        pr = q[0]; pi = q[1]; qr = q[4]; qi = q[5];   // P_{s-1} * (1,0)^T
    }
    float* Or = isR ? (Rr + b * Nq) : (Lr + b * Nq);
    float* Oi = isR ? (Ri + b * Nq) : (Li + b * Nq);
    for (int j = 0; j < 32; ++j) {
        const int p_  = seg * 32 + j;
        const int idx = isR ? (Nq - 1 - p_) : p_;
        const float er = vb[idx] - 2.0f, ei = -1.0f;
        const float npr = er * pr - ei * pi - qr;      // p' = e*p - q ; q' = p
        const float npi = er * pi + ei * pr - qi;
        qr = pr; qi = pi; pr = npr; pi = npi;
        const float den = qr * qr + qi * qi;
        const float inv = __builtin_amdgcn_rcpf(den);
        Or[idx] = (pr * qr + pi * qi) * inv;           // p/q = p*conj(q)/|q|^2
        Oi[idx] = (pi * qr - pr * qi) * inv;
        const float mx = fmaxf(fmaxf(fabsf(pr), fabsf(pi)), fmaxf(fabsf(qr), fabsf(qi)));
        const float s = __builtin_amdgcn_rcpf(fmaxf(mx, 1e-30f));
        pr *= s; pi *= s; qr *= s; qi *= s;
    }
}

// ---------------- kernel 6: Green fn + rank-2 spec, in-place add ----------------
__global__ __launch_bounds__(256) void final_kernel(
    float* __restrict__ out, const float* __restrict__ v,
    const float* __restrict__ Lr, const float* __restrict__ Li,
    const float* __restrict__ Rr, const float* __restrict__ Ri,
    const float* __restrict__ out_w, const float* __restrict__ out_b,
    const float* __restrict__ bk) {
    const size_t i = (size_t)blockIdx.x * 256 + threadIdx.x;
    if (i >= (size_t)T * Dq) return;
    const int tok = (int)(i / Dq);
    const int c   = (int)(i % Dq);
    const float dr   = v[tok] - 2.0f;              // di = -1
    const float denr = Lr[tok] + Rr[tok] - dr;
    const float deni = Li[tok] + Ri[tok] + 1.0f;
    const float m  = denr * denr + deni * deni;
    float gr = denr / m, gi = -deni / m;
    gr = fminf(fmaxf(gr, -FEAT_CLAMP), FEAT_CLAMP);
    gi = fminf(fmaxf(gi, -FEAT_CLAMP), FEAT_CLAMP);
    const float spec = gr * out_w[c] + gi * out_w[Dq + c] + out_b[c];
    out[i] += bk[0] * spec;
}

// ---------------- launch ----------------
extern "C" void kernel_launch(void* const* d_in, const int* in_sizes, int n_in,
                              void* d_out, int out_size, void* d_ws, size_t ws_size,
                              hipStream_t stream) {
    const float* x      = (const float*)d_in[0];
    const float* gate_w = (const float*)d_in[1];
    const float* gate_b = (const float*)d_in[2];
    const float* w1     = (const float*)d_in[3];
    const float* b1     = (const float*)d_in[4];
    const float* w2     = (const float*)d_in[5];
    const float* b2     = (const float*)d_in[6];
    const float* v_w    = (const float*)d_in[7];
    const float* v_b    = (const float*)d_in[8];
    const float* out_w  = (const float*)d_in[9];
    const float* out_b  = (const float*)d_in[10];
    const float* bk     = (const float*)d_in[11];
    float* out = (float*)d_out;

    // workspace layout (256B aligned slots)
    char* ws = (char*)d_ws;
    constexpr size_t SZ_W1T   = (size_t)Eq * Hq * Dq * sizeof(_Float16);  // 8 MiB
    constexpr size_t SZ_W2T   = (size_t)Eq * Dq * Hq * sizeof(_Float16);  // 8 MiB
    constexpr size_t SZ_XH    = (size_t)T * Dq * sizeof(_Float16);        // 16 MiB
    constexpr size_t SZ_LISTS = (size_t)Eq * T * sizeof(int);             // 256 KiB
    constexpr size_t SZ_CNT   = 256;
    constexpr size_t SZ_VEC   = (size_t)T * sizeof(float);                // 64 KiB
    size_t off = 0;
    _Float16* W1T  = (_Float16*)(ws + off); off += SZ_W1T;
    _Float16* W2T  = (_Float16*)(ws + off); off += SZ_W2T;
    _Float16* XH   = (_Float16*)(ws + off); off += SZ_XH;
    int*      lists= (int*)(ws + off);      off += SZ_LISTS;
    int*      cnts = (int*)(ws + off);      off += SZ_CNT;
    float*    vbuf = (float*)(ws + off);    off += SZ_VEC;
    float*    Lr   = (float*)(ws + off);    off += SZ_VEC;
    float*    Li   = (float*)(ws + off);    off += SZ_VEC;
    float*    Rr   = (float*)(ws + off);    off += SZ_VEC;
    float*    Ri   = (float*)(ws + off);    off += SZ_VEC;

    zero_counts_kernel<<<1, 32, 0, stream>>>(cnts);

    // W1 (E,D,H) -> W1T [E][H][D];  W2 (E,H,D) -> W2T [E][D][H]
    transpose_cvt_kernel<<<dim3(Hq / 32, Dq / 32, Eq), dim3(32, 8), 0, stream>>>(w1, W1T, Dq, Hq);
    transpose_cvt_kernel<<<dim3(Dq / 32, Hq / 32, Eq), dim3(32, 8), 0, stream>>>(w2, W2T, Hq, Dq);

    cvt_x_kernel<<<(int)(((size_t)T * Dq) / 8 / 256), 256, 0, stream>>>(x, XH);

    gate_kernel<<<(T * 32) / 256, 256, 0, stream>>>(x, gate_w, gate_b, lists, cnts);

    ffn_kernel<<<dim3(T / MT, Eq), 256, 0, stream>>>(XH, W1T, b1, W2T, b2, lists, cnts, out);

    v_kernel<<<(T * 32) / 256, 256, 0, stream>>>(out, v_w, v_b, vbuf);

    tridiag_scan_kernel<<<1, 1024, 0, stream>>>(vbuf, Lr, Li, Rr, Ri);

    final_kernel<<<(int)(((size_t)T * Dq) / 256), 256, 0, stream>>>(
        out, vbuf, Lr, Li, Rr, Ri, out_w, out_b, bk);
}
